// MambaLayer_8985071583799
// MI455X (gfx1250) — matrix operand order repaired and checked by reading, not compiled, not run on hardware
//
#include <hip/hip_runtime.h>
#include <hip/hip_bf16.h>

// ---------------------------------------------------------------------------
// Types for CDNA5 WMMA (wave32)
// ---------------------------------------------------------------------------
typedef __attribute__((ext_vector_type(16))) __bf16 v16bf;
typedef __attribute__((ext_vector_type(8)))  __bf16 v8bf;
typedef __attribute__((ext_vector_type(8)))  float  v8f;

union FragBF { v16bf v; v8bf h[2]; };

// Async global->LDS path (gfx1250 ASYNCcnt). Guarded: falls back to a
// register-bounce copy (ds_store_b128) if the builtins are not declared.
#if defined(__has_builtin)
#  if __has_builtin(__builtin_amdgcn_global_load_async_to_lds_b128) && \
      __has_builtin(__builtin_amdgcn_s_wait_asynccnt)
#    define HAVE_ASYNC_LDS 1
#  endif
#endif
#ifndef HAVE_ASYNC_LDS
#  define HAVE_ASYNC_LDS 0
#endif

#if HAVE_ASYNC_LDS
// Builtin signature (from clang diagnostic): (v4i AS1*, v4i AS3*, imm i32, imm i32)
typedef int v4i_t __attribute__((ext_vector_type(4)));
typedef __attribute__((address_space(1))) v4i_t as1_v4i;
typedef __attribute__((address_space(3))) v4i_t as3_v4i;
#endif

__device__ __forceinline__ int imin(int a, int b) { return a < b ? a : b; }

__device__ __forceinline__ __bf16 f2bf(float f) {
    union { float f; unsigned u; } a; a.f = f;
    unsigned r = a.u + 0x7FFFu + ((a.u >> 16) & 1u);   // round-to-nearest-even
    union { unsigned short s; __bf16 b; } o;
    o.s = (unsigned short)(r >> 16);
    return o.b;
}

__device__ __forceinline__ float silu_f(float x) { return x / (1.0f + __expf(-x)); }

// Copy 32 bytes global -> LDS (async if available).
__device__ __forceinline__ void cp32B_g2l(void* l, const void* g) {
#if HAVE_ASYNC_LDS
    as1_v4i* gp = (as1_v4i*)g;
    as3_v4i* lp = (as3_v4i*)l;
    __builtin_amdgcn_global_load_async_to_lds_b128(gp, lp, 0, 0);
    __builtin_amdgcn_global_load_async_to_lds_b128(gp, lp, 16, 0);
#else
    ((v8bf*)l)[0] = ((const v8bf*)g)[0];
    ((v8bf*)l)[1] = ((const v8bf*)g)[1];
#endif
}

__device__ __forceinline__ void async_wait_all() {
#if HAVE_ASYNC_LDS
    __builtin_amdgcn_s_wait_asynccnt(0);
#endif
}

// ---------------------------------------------------------------------------
// WMMA fragment helpers per CDNA5 ISA 7.12.2 (16-bit A 16x32, B 32x16, f32 C/D)
// ---------------------------------------------------------------------------
__device__ __forceinline__ v16bf load_a_frag(const __bf16* __restrict__ A, int lda,
                                             int tileM, int k0, int lane) {
    int r = lane & 15, hi = lane >> 4;
    const __bf16* p = A + (size_t)(tileM + r) * lda + k0 + 8 * hi;
    FragBF f;
    f.h[0] = *(const v8bf*)(p);        // K = k0+8*hi   .. +7
    f.h[1] = *(const v8bf*)(p + 16);   // K = k0+16+8*hi.. +7
    return f.v;
}

__device__ __forceinline__ v16bf load_b_frag(const __bf16* __restrict__ Bt, int ldb,
                                             int tileN, int k0, int lane) {
    // Bt is [K, N] row-major: lane holds column N = tileN + lane%16, K = k0+8*hi+s / k0+16+8*hi+s
    int c = lane & 15, hi = lane >> 4;
    const __bf16* p = Bt + (size_t)(k0 + 8 * hi) * ldb + tileN + c;
    v16bf v;
#pragma unroll
    for (int i = 0; i < 8; ++i) {
        v[i]     = p[(size_t)i * ldb];
        v[8 + i] = p[(size_t)(16 + i) * ldb];
    }
    return v;
}

// LDS-resident fragments (sA: [64][32] row-major, sB: [32][64] row-major)
__device__ __forceinline__ v16bf lds_a_frag(const __bf16* sA, int m0, int lane) {
    int r = lane & 15, hi = lane >> 4;
    const __bf16* p = sA + (m0 + r) * 32 + 8 * hi;
    FragBF f;
    f.h[0] = *(const v8bf*)(p);
    f.h[1] = *(const v8bf*)(p + 16);
    return f.v;
}

__device__ __forceinline__ v16bf lds_b_frag(const __bf16* sB, int n0, int lane) {
    int c = lane & 15, hi = lane >> 4;
    const __bf16* p = sB + (8 * hi) * 64 + n0 + c;
    v16bf v;
#pragma unroll
    for (int i = 0; i < 8; ++i) {
        v[i]     = p[i * 64];
        v[8 + i] = p[(16 + i) * 64];
    }
    return v;
}

__device__ __forceinline__ void store_tile(float* __restrict__ C, int ldc,
                                           int tileM, int tileN, int lane, v8f acc) {
    int col = lane & 15, hi = lane >> 4;
#pragma unroll
    for (int r = 0; r < 8; ++r)
        C[(size_t)(tileM + r + 8 * hi) * ldc + tileN + col] = acc[r];
}

#define WMMA_BF16(a, b, c) \
    __builtin_amdgcn_wmma_f32_16x16x32_bf16(false, (a), false, (b), (short)0, (c), false, false)

// ---------------------------------------------------------------------------
// LDS-staged WMMA GEMM (requires M%64==0, N%64==0, K%32==0).
// 4 waves / workgroup -> 64x64 C tile; double-buffered async global->LDS.
// ---------------------------------------------------------------------------
__global__ __launch_bounds__(128)
void wmma_gemm_lds(const __bf16* __restrict__ A, const __bf16* __restrict__ Bt,
                   float* __restrict__ C, int M, int N, int K, int lda) {
    __shared__ __align__(16) __bf16 lA[2][64 * 32];   // [buf][row*32 + k]
    __shared__ __align__(16) __bf16 lB[2][32 * 64];   // [buf][k*64 + n]

    const int tid   = threadIdx.x;
    const int lane  = tid & 31;
    const int wave  = tid >> 5;
    const int wm    = (wave >> 1) * 32;   // wave quadrant in 64x64 tile
    const int wn    = (wave & 1) * 32;
    const int tileM = blockIdx.y * 64;
    const int tileN = blockIdx.x * 64;

    // Per-thread cooperative staging coordinates (128 threads, 32B each op)
    const int arow = tid >> 1, ak = (tid & 1) * 16;   // A: 64 rows x 32 k
    const int brow = tid >> 2, bn = (tid & 3) * 16;   // B: 32 k   x 64 n

    v8f acc00 = {}, acc01 = {}, acc10 = {}, acc11 = {};

    const int ksteps = K >> 5;

    // Prologue: stage K-step 0 into buffer 0
    cp32B_g2l(&lA[0][arow * 32 + ak], A + (size_t)(tileM + arow) * lda + ak);
    cp32B_g2l(&lB[0][brow * 64 + bn], Bt + (size_t)brow * N + tileN + bn);

    for (int kt = 0; kt < ksteps; ++kt) {
        async_wait_all();
        __syncthreads();                   // current buffer published

        if (kt + 1 < ksteps) {             // stage next K-step into other buffer
            int k1 = (kt + 1) << 5, nb = (kt + 1) & 1;
            cp32B_g2l(&lA[nb][arow * 32 + ak],
                      A + (size_t)(tileM + arow) * lda + k1 + ak);
            cp32B_g2l(&lB[nb][brow * 64 + bn],
                      Bt + (size_t)(k1 + brow) * N + tileN + bn);
        }

        const __bf16* sA = &lA[kt & 1][0];
        const __bf16* sB = &lB[kt & 1][0];
        v16bf a0 = lds_a_frag(sA, wm,      lane);
        v16bf a1 = lds_a_frag(sA, wm + 16, lane);
        v16bf b0 = lds_b_frag(sB, wn,      lane);
        v16bf b1 = lds_b_frag(sB, wn + 16, lane);
        acc00 = WMMA_BF16(a0, b0, acc00);
        acc01 = WMMA_BF16(a0, b1, acc01);
        acc10 = WMMA_BF16(a1, b0, acc10);
        acc11 = WMMA_BF16(a1, b1, acc11);
    }

    store_tile(C, N, tileM + wm,      tileN + wn,      lane, acc00);
    store_tile(C, N, tileM + wm,      tileN + wn + 16, lane, acc01);
    store_tile(C, N, tileM + wm + 16, tileN + wn,      lane, acc10);
    store_tile(C, N, tileM + wm + 16, tileN + wn + 16, lane, acc11);
}

// ---------------------------------------------------------------------------
// Direct (no-LDS) WMMA GEMM for small N (x_proj, N=96): one wave / 32x32 tile.
// ---------------------------------------------------------------------------
__global__ __launch_bounds__(32)
void wmma_gemm_bf16(const __bf16* __restrict__ A, const __bf16* __restrict__ Bt,
                    float* __restrict__ C, int M, int N, int K, int lda) {
    int tileN = blockIdx.x * 32;
    int tileM = blockIdx.y * 32;
    int lane  = threadIdx.x;
    v8f acc00 = {}, acc01 = {}, acc10 = {}, acc11 = {};

    for (int k0 = 0; k0 < K; k0 += 32) {
        if (k0 + 32 < K) {
            __builtin_prefetch(A + (size_t)(tileM + (lane & 15)) * lda + k0 + 32, 0, 3);
            __builtin_prefetch(Bt + (size_t)(k0 + 32 + lane) * N + tileN, 0, 3);
        }
        v16bf a0 = load_a_frag(A, lda, tileM,      k0, lane);
        v16bf a1 = load_a_frag(A, lda, tileM + 16, k0, lane);
        v16bf b0 = load_b_frag(Bt, N,  tileN,      k0, lane);
        v16bf b1 = load_b_frag(Bt, N,  tileN + 16, k0, lane);
        acc00 = WMMA_BF16(a0, b0, acc00);
        acc01 = WMMA_BF16(a0, b1, acc01);
        acc10 = WMMA_BF16(a1, b0, acc10);
        acc11 = WMMA_BF16(a1, b1, acc11);
    }
    store_tile(C, N, tileM,      tileN,      lane, acc00);
    store_tile(C, N, tileM,      tileN + 16, lane, acc01);
    store_tile(C, N, tileM + 16, tileN,      lane, acc10);
    store_tile(C, N, tileM + 16, tileN + 16, lane, acc11);
}

// ---------------------------------------------------------------------------
// Elementwise converts
// ---------------------------------------------------------------------------
__global__ void cvt_f32_bf16(const float* __restrict__ src, __bf16* __restrict__ dst, int n) {
    int t = blockIdx.x * 256 + threadIdx.x;
    if (t < n) dst[t] = f2bf(src[t]);
}

// W[N,K] f32 -> Wt[K,N] bf16
__global__ void transpose_cvt(const float* __restrict__ W, __bf16* __restrict__ Wt,
                              int N, int K) {
    int t = blockIdx.x * 256 + threadIdx.x;
    if (t >= N * K) return;
    int n = t / K, k = t % K;
    Wt[(size_t)k * N + n] = f2bf(W[t]);
}

// ---------------------------------------------------------------------------
// PPM kernels.  x: [2,512,32,32], seq: [2048, 1024]
// ---------------------------------------------------------------------------
__global__ void pool0_kernel(const float* __restrict__ x, const float* __restrict__ pool_w,
                             const float* __restrict__ g, const float* __restrict__ be,
                             const float* __restrict__ mu, const float* __restrict__ va,
                             float* __restrict__ y0) {
    int bo = blockIdx.x;                 // 0..255
    int b = bo >> 7, o = bo & 127;
    const float* wrow = pool_w + (size_t)o * 512;   // layer 0
    float inv = rsqrtf(va[o] + 1e-5f);
    float sc = g[o] * inv, m0 = mu[o], bb = be[o];
    float acc = 0.f;
    for (int p = threadIdx.x; p < 1024; p += 256) {
        const float* xp = x + (size_t)b * 512 * 1024 + p;
        float s = 0.f;
        for (int c = 0; c < 512; ++c) s += wrow[c] * xp[(size_t)c * 1024];
        float v = (s - m0) * sc + bb;
        v = fminf(fmaxf(v, 0.f), 6.f);
        acc += v;
    }
    __shared__ float red[256];
    red[threadIdx.x] = acc;
    __syncthreads();
    for (int st = 128; st > 0; st >>= 1) {
        if ((int)threadIdx.x < st) red[threadIdx.x] += red[threadIdx.x + st];
        __syncthreads();
    }
    if (threadIdx.x == 0) y0[bo] = red[0] * (1.f / 1024.f);
}

__global__ void write_seq_x(const float* __restrict__ x, float* __restrict__ seq) {
    int t = blockIdx.x * 256 + threadIdx.x;          // B*512*1024 = 1048576
    if (t >= 2 * 512 * 1024) return;
    int p = t & 1023, c = (t >> 10) & 511, b = t >> 19;
    seq[((size_t)b * 1024 + p) * 1024 + c] = x[((size_t)b * 512 + c) * 1024 + p];
}

__global__ void write_seq_y0(const float* __restrict__ y0, float* __restrict__ seq) {
    int t = blockIdx.x * 256 + threadIdx.x;          // B*1024*128 = 262144
    if (t >= 2 * 1024 * 128) return;
    int o = t & 127, p = (t >> 7) & 1023, b = t >> 17;
    seq[((size_t)b * 1024 + p) * 1024 + 512 + o] = y0[b * 128 + o];
}

__global__ void pool_scale(const float* __restrict__ x, float* __restrict__ pooled, int s) {
    int t = blockIdx.x * 256 + threadIdx.x;
    int total = 2 * 512 * s * s;
    if (t >= total) return;
    int q = t % s, p = (t / s) % s, c = (t / (s * s)) % 512, b = t / (512 * s * s);
    int ps = p * 32 / s, pe = ((p + 1) * 32 + s - 1) / s;
    int qs = q * 32 / s, qe = ((q + 1) * 32 + s - 1) / s;
    const float* xp = x + ((size_t)b * 512 + c) * 1024;
    float sum = 0.f;
    for (int h = ps; h < pe; ++h)
        for (int w = qs; w < qe; ++w)
            sum += xp[h * 32 + w];
    pooled[t] = sum / (float)((pe - ps) * (qe - qs));
}

__global__ void conv_bn_scale(const float* __restrict__ pooled, const float* __restrict__ pool_w,
                              const float* __restrict__ g, const float* __restrict__ be,
                              const float* __restrict__ mu, const float* __restrict__ va,
                              float* __restrict__ ys, int s, int i) {
    int t = blockIdx.x * 256 + threadIdx.x;
    int total = 2 * 128 * s * s;
    if (t >= total) return;
    int pq = t % (s * s), o = (t / (s * s)) % 128, b = t / (128 * s * s);
    const float* w  = pool_w + ((size_t)i * 128 + o) * 512;
    const float* pp = pooled + (size_t)b * 512 * s * s + pq;
    float sum = 0.f;
    for (int c = 0; c < 512; ++c) sum += w[c] * pp[(size_t)c * s * s];
    float inv = rsqrtf(va[i * 128 + o] + 1e-5f);
    float v = (sum - mu[i * 128 + o]) * (g[i * 128 + o] * inv) + be[i * 128 + o];
    ys[t] = fminf(fmaxf(v, 0.f), 6.f);
}

__global__ void upsample_scale(const float* __restrict__ ys, float* __restrict__ seq,
                               int s, int i) {
    int t = blockIdx.x * 256 + threadIdx.x;          // 2*128*1024
    if (t >= 2 * 128 * 1024) return;
    int w = t & 31, h = (t >> 5) & 31, o = (t >> 10) & 127, b = t >> 17;
    float sh = (h + 0.5f) * s / 32.f - 0.5f;
    sh = fminf(fmaxf(sh, 0.f), (float)(s - 1));
    int h0 = (int)sh, h1 = imin(h0 + 1, s - 1); float wh = sh - h0;
    float sw = (w + 0.5f) * s / 32.f - 0.5f;
    sw = fminf(fmaxf(sw, 0.f), (float)(s - 1));
    int w0 = (int)sw, w1 = imin(w0 + 1, s - 1); float ww = sw - w0;
    const float* yb = ys + ((size_t)b * 128 + o) * s * s;
    float v = (1.f - wh) * ((1.f - ww) * yb[h0 * s + w0] + ww * yb[h0 * s + w1])
            +        wh  * ((1.f - ww) * yb[h1 * s + w0] + ww * yb[h1 * s + w1]);
    seq[((size_t)b * 1024 + h * 32 + w) * 1024 + 512 + 128 * i + o] = v;
}

// ---------------------------------------------------------------------------
// Mamba pointwise kernels
// ---------------------------------------------------------------------------
__global__ void conv1d_silu(const float* __restrict__ xz, const float* __restrict__ cw,
                            const float* __restrict__ cb, float* __restrict__ uf,
                            __bf16* __restrict__ ub) {
    int t = blockIdx.x * 256 + threadIdx.x;          // 2048*2048
    if (t >= 2048 * 2048) return;
    int d = t & 2047, m = t >> 11;
    int b = m >> 10, l = m & 1023;
    float acc = cb[d];
#pragma unroll
    for (int tt = 0; tt < 4; ++tt) {
        int ll = l - 3 + tt;
        if (ll >= 0)
            acc += cw[d * 4 + tt] * xz[((size_t)b * 1024 + ll) * 4096 + d];
    }
    float u = silu_f(acc);
    uf[t] = u;
    ub[t] = f2bf(u);
}

__global__ void bias_softplus(float* __restrict__ delta, const float* __restrict__ dtb) {
    int t = blockIdx.x * 256 + threadIdx.x;
    if (t >= 2048 * 2048) return;
    float v = delta[t] + dtb[t & 2047];
    delta[t] = (v > 20.f) ? v : log1pf(__expf(v));
}

// Selective scan: thread = (b, d), 16 SSM states in registers, sequential in L.
__global__ void scan_kernel(const float* __restrict__ delta, const float* __restrict__ u,
                            const float* __restrict__ xdbl, const float* __restrict__ xz,
                            const float* __restrict__ A_log, const float* __restrict__ Dv,
                            float* __restrict__ yf, __bf16* __restrict__ yb) {
    int t = blockIdx.x * 256 + threadIdx.x;
    if (t >= 4096) return;
    int b = t >> 11, d = t & 2047;
    float Aa[16], h[16];
#pragma unroll
    for (int n = 0; n < 16; ++n) { Aa[n] = -__expf(A_log[d * 16 + n]); h[n] = 0.f; }
    float Dd = Dv[d];
    for (int l = 0; l < 1024; ++l) {
        size_t m = (size_t)b * 1024 + l;
        float dv = delta[m * 2048 + d];
        float uv = u[m * 2048 + d];
        const float* xd = xdbl + m * 96;
        float du = dv * uv;
        float y = 0.f;
#pragma unroll
        for (int n = 0; n < 16; ++n) {
            h[n] = __expf(dv * Aa[n]) * h[n] + du * xd[64 + n];
            y += h[n] * xd[80 + n];
        }
        y += uv * Dd;
        float z = xz[m * 4096 + 2048 + d];
        y *= silu_f(z);
        yf[m * 2048 + d] = y;
        yb[m * 2048 + d] = f2bf(y);
    }
}

// outC [2048,1024] -> d_out [2,1024,32,32]
__global__ void final_transpose(const float* __restrict__ outC, float* __restrict__ out) {
    int t = blockIdx.x * 256 + threadIdx.x;          // 2048*1024
    if (t >= 2048 * 1024) return;
    int n = t & 1023, m = t >> 10, b = m >> 10, p = m & 1023;
    out[((size_t)b * 1024 + n) * 1024 + p] = outC[(size_t)m * 1024 + n];
}

// ---------------------------------------------------------------------------
// Host launcher
// ---------------------------------------------------------------------------
extern "C" void kernel_launch(void* const* d_in, const int* in_sizes, int n_in,
                              void* d_out, int out_size, void* d_ws, size_t ws_size,
                              hipStream_t stream) {
    const float* x          = (const float*)d_in[0];
    const float* pool_w     = (const float*)d_in[1];
    const float* bn_g       = (const float*)d_in[2];
    const float* bn_b       = (const float*)d_in[3];
    const float* bn_m       = (const float*)d_in[4];
    const float* bn_v       = (const float*)d_in[5];
    const float* in_proj_w  = (const float*)d_in[6];
    const float* conv_w     = (const float*)d_in[7];
    const float* conv_b     = (const float*)d_in[8];
    const float* x_proj_w   = (const float*)d_in[9];
    const float* dt_proj_w  = (const float*)d_in[10];
    const float* dt_proj_b  = (const float*)d_in[11];
    const float* A_log      = (const float*)d_in[12];
    const float* Dv         = (const float*)d_in[13];
    const float* out_proj_w = (const float*)d_in[14];
    float* out = (float*)d_out;
    (void)in_sizes; (void)n_in; (void)out_size; (void)ws_size;

    char* base = (char*)d_ws;
    size_t off = 0;
    auto take = [&](size_t bytes) -> char* {
        char* p = base + off;
        off += (bytes + 255) & ~(size_t)255;
        return p;
    };
    float*  seq     = (float*) take(2048ull * 1024 * 4);
    __bf16* seq_bf  = (__bf16*)take(2048ull * 1024 * 2);
    float*  y0      = (float*) take(256 * 4);
    float*  pooled  = (float*) take(2ull * 512 * 625 * 4);
    float*  yscale  = (float*) take(2ull * 128 * 625 * 4);
    __bf16* wt1     = (__bf16*)take(1024ull * 4096 * 2);
    float*  xz      = (float*) take(2048ull * 4096 * 4);
    float*  u_f     = (float*) take(2048ull * 2048 * 4);
    __bf16* u_bf    = (__bf16*)take(2048ull * 2048 * 2);
    __bf16* wt2     = (__bf16*)take(2048ull * 96 * 2);
    float*  xdbl    = (float*) take(2048ull * 96 * 4);
    __bf16* xdbl_bf = (__bf16*)take(2048ull * 96 * 2);
    __bf16* wt3     = (__bf16*)take(64ull * 2048 * 2);
    float*  delta   = (float*) take(2048ull * 2048 * 4);
    float*  y_f     = (float*) take(2048ull * 2048 * 4);
    __bf16* y_bf    = (__bf16*)take(2048ull * 2048 * 2);
    __bf16* wt4     = (__bf16*)take(2048ull * 1024 * 2);
    float*  outC    = xz;   // xz region is dead after scan_kernel; reuse

    auto blocks = [](long long n) { return (unsigned)((n + 255) / 256); };

    // ---- PPM ----
    pool0_kernel<<<256, 256, 0, stream>>>(x, pool_w, bn_g, bn_b, bn_m, bn_v, y0);
    write_seq_x<<<blocks(2ll * 512 * 1024), 256, 0, stream>>>(x, seq);
    write_seq_y0<<<blocks(2ll * 1024 * 128), 256, 0, stream>>>(y0, seq);
    const int scales[3] = {9, 17, 25};
    for (int i = 1; i <= 3; ++i) {
        int s = scales[i - 1];
        pool_scale<<<blocks(2ll * 512 * s * s), 256, 0, stream>>>(x, pooled, s);
        conv_bn_scale<<<blocks(2ll * 128 * s * s), 256, 0, stream>>>(
            pooled, pool_w, bn_g, bn_b, bn_m, bn_v, yscale, s, i);
        upsample_scale<<<blocks(2ll * 128 * 1024), 256, 0, stream>>>(yscale, seq, s, i);
    }

    // ---- in_proj GEMM: xz[2048,4096] = seq[2048,1024] @ in_proj_w^T ----
    cvt_f32_bf16<<<blocks(2048ll * 1024), 256, 0, stream>>>(seq, seq_bf, 2048 * 1024);
    transpose_cvt<<<blocks(4096ll * 1024), 256, 0, stream>>>(in_proj_w, wt1, 4096, 1024);
    wmma_gemm_lds<<<dim3(4096 / 64, 2048 / 64), 128, 0, stream>>>(
        seq_bf, wt1, xz, 2048, 4096, 1024, 1024);

    // ---- causal depthwise conv1d + SiLU ----
    conv1d_silu<<<blocks(2048ll * 2048), 256, 0, stream>>>(xz, conv_w, conv_b, u_f, u_bf);

    // ---- x_proj GEMM: xdbl[2048,96] = u[2048,2048] @ x_proj_w^T (N=96: direct) ----
    transpose_cvt<<<blocks(96ll * 2048), 256, 0, stream>>>(x_proj_w, wt2, 96, 2048);
    wmma_gemm_bf16<<<dim3(96 / 32, 2048 / 32), 32, 0, stream>>>(
        u_bf, wt2, xdbl, 2048, 96, 2048, 2048);
    cvt_f32_bf16<<<blocks(2048ll * 96), 256, 0, stream>>>(xdbl, xdbl_bf, 2048 * 96);

    // ---- dt_proj GEMM: delta_raw[2048,2048] = xdbl[:, :64] @ dt_proj_w^T ----
    transpose_cvt<<<blocks(2048ll * 64), 256, 0, stream>>>(dt_proj_w, wt3, 2048, 64);
    wmma_gemm_lds<<<dim3(2048 / 64, 2048 / 64), 128, 0, stream>>>(
        xdbl_bf, wt3, delta, 2048, 2048, 64, 96);
    bias_softplus<<<blocks(2048ll * 2048), 256, 0, stream>>>(delta, dt_proj_b);

    // ---- selective scan ----
    scan_kernel<<<16, 256, 0, stream>>>(delta, u_f, xdbl, xz, A_log, Dv, y_f, y_bf);

    // ---- out_proj GEMM: outC[2048,1024] = y[2048,2048] @ out_proj_w^T ----
    transpose_cvt<<<blocks(1024ll * 2048), 256, 0, stream>>>(out_proj_w, wt4, 1024, 2048);
    wmma_gemm_lds<<<dim3(1024 / 64, 2048 / 64), 128, 0, stream>>>(
        y_bf, wt4, outC, 2048, 1024, 2048, 2048);

    // ---- final layout [B, D_MODEL, H, W] ----
    final_transpose<<<blocks(2048ll * 1024), 256, 0, stream>>>(outC, out);
}
